// MetaController_73452530696558
// MI455X (gfx1250) — matrix operand order repaired and checked by reading, not compiled, not run on hardware
//
#include <hip/hip_runtime.h>

#define B_    64
#define T_    1024
#define NE_   256
#define NZ_   64
#define H_    256
#define SWH_  128
#define BT_   (B_ * T_)      // 65536
#define G3H_  (3 * H_)       // 768
#define H2_   (2 * H_)       // 512
#define W1COLS_ 832          // NE + 2H + NZ

typedef __attribute__((ext_vector_type(16))) __bf16 v16bf;
typedef __attribute__((ext_vector_type(8)))  float  v8f;

union FragU { uint4 u[2]; v16bf v; };

__device__ __forceinline__ unsigned short f2bfu(float f) {
    __bf16 h = (__bf16)f;
    return __builtin_bit_cast(unsigned short, h);
}

// ---------------------------------------------------------------------------
// fp32 -> bf16 bulk convert
// ---------------------------------------------------------------------------
__global__ void k_cvt_bf16(const float* __restrict__ x,
                           unsigned short* __restrict__ y, long n) {
    long i = (long)blockIdx.x * blockDim.x + threadIdx.x;
    if (i < n) y[i] = f2bfu(x[i]);
}

// ---------------------------------------------------------------------------
// Pack a weight matrix W[N][K] (row n = output column n, optionally a column
// slice of a wider matrix) into WMMA B-fragment order:
//   dst[((nt*KT)+kt)*512 + lane*16 + e] = bf16( W[nt*16 + lane%16]
//                                                [kt*32 + (lane<16?0:8) + (e<8?e:8+e)] )
// so a fragment load is one contiguous 32B read per lane.
// ---------------------------------------------------------------------------
__global__ void k_pack_b(const float* __restrict__ W, int stride, int colOff,
                         int N, int K, unsigned short* __restrict__ dst) {
    int  KT    = K >> 5;
    long total = (long)(N >> 4) * KT * 512;
    long i     = (long)blockIdx.x * blockDim.x + threadIdx.x;
    if (i >= total) return;
    int  slot = (int)(i & 511);
    long tile = i >> 9;
    int  kt   = (int)(tile % KT);
    int  nt   = (int)(tile / KT);
    int  lane = slot >> 4, e = slot & 15;
    int  n    = nt * 16 + (lane & 15);
    int  k    = kt * 32 + ((lane < 16) ? 0 : 8) + ((e < 8) ? e : 8 + e);
    dst[i]    = f2bfu(W[(long)n * stride + colOff + k]);
}

__global__ void k_addvec(const float* __restrict__ a, const float* __restrict__ b,
                         float* __restrict__ o, int n) {
    int i = blockIdx.x * blockDim.x + threadIdx.x;
    if (i < n) o[i] = a[i] + b[i];
}

// ---------------------------------------------------------------------------
// Generic bf16 WMMA GEMM:  out[M=BT_][N] = A[M][K] @ packedB + bias + C0
// One wave computes a 16x64 output tile (4 n-tiles), K-loop of 32.
// ---------------------------------------------------------------------------
__global__ __launch_bounds__(256) void k_gemm(
        const unsigned short* __restrict__ A,
        const unsigned short* __restrict__ pB,
        const float* __restrict__ bias,
        const float* C0, float* out, int N, int K) {
    const int lane = threadIdx.x & 31;
    const int gw   = blockIdx.x * (blockDim.x >> 5) + (threadIdx.x >> 5);
    const int ngr  = N >> 6;
    const int mt   = gw / ngr;
    const int ng   = gw % ngr;
    if (mt >= (BT_ >> 4)) return;

    const int mrow = mt * 16 + (lane & 15);
    const int kb   = (lane < 16) ? 0 : 8;
    const int KT   = K >> 5;
    const int nt0  = ng * 4;

    v8f acc0 = {0,0,0,0,0,0,0,0}, acc1 = acc0, acc2 = acc0, acc3 = acc0;

    for (int kt = 0; kt < KT; ++kt) {
        FragU a;
        const unsigned short* pa = A + (long)mrow * K + kt * 32 + kb;
        __builtin_prefetch(pa + 64, 0, 1);
        a.u[0] = *(const uint4*)(pa);        // K = kb .. kb+7
        a.u[1] = *(const uint4*)(pa + 16);   // K = kb+16 .. kb+23

        const unsigned short* pb =
            pB + ((long)nt0 * KT + kt) * 512 + lane * 16;
        FragU b0, b1, b2, b3;
        b0.u[0] = *(const uint4*)(pb);                 b0.u[1] = *(const uint4*)(pb + 8);
        b1.u[0] = *(const uint4*)(pb + (long)KT*512);      b1.u[1] = *(const uint4*)(pb + (long)KT*512 + 8);
        b2.u[0] = *(const uint4*)(pb + (long)2*KT*512);    b2.u[1] = *(const uint4*)(pb + (long)2*KT*512 + 8);
        b3.u[0] = *(const uint4*)(pb + (long)3*KT*512);    b3.u[1] = *(const uint4*)(pb + (long)3*KT*512 + 8);

        acc0 = __builtin_amdgcn_wmma_f32_16x16x32_bf16(false, a.v, false, b0.v, (short)0, acc0, false, false);
        acc1 = __builtin_amdgcn_wmma_f32_16x16x32_bf16(false, a.v, false, b1.v, (short)0, acc1, false, false);
        acc2 = __builtin_amdgcn_wmma_f32_16x16x32_bf16(false, a.v, false, b2.v, (short)0, acc2, false, false);
        acc3 = __builtin_amdgcn_wmma_f32_16x16x32_bf16(false, a.v, false, b3.v, (short)0, acc3, false, false);
    }

    const int rbase = mt * 16 + ((lane >> 4) << 3);
    v8f accs[4] = {acc0, acc1, acc2, acc3};
#pragma unroll
    for (int q = 0; q < 4; ++q) {
        const int col = (nt0 + q) * 16 + (lane & 15);
        const float bv = bias ? bias[col] : 0.f;
#pragma unroll
        for (int v = 0; v < 8; ++v) {
            long idx = (long)(rbase + v) * N + col;
            float r = accs[q][v] + bv;
            if (C0) r += C0[idx];
            out[idx] = r;
        }
    }
}

// ---------------------------------------------------------------------------
// Persistent GRU recurrence: one workgroup per direction, 16 waves.
// LDS: hbf (bf16 image of h, 64x256, 32KB) + hp (fp32 64x768, 192KB).
// Per step: hp = h_bf16 @ w_hh^T (WMMA), then fused gate math; per-element
// fp32 h state lives in registers (fixed thread->element mapping).
// ---------------------------------------------------------------------------
#define GRU_LDS (B_ * G3H_ * 4 + B_ * H_ * 2)

__global__ __launch_bounds__(512, 1) void k_gru(
        const float* __restrict__ xp0, const float* __restrict__ xp1,
        const unsigned short* __restrict__ pW0,
        const unsigned short* __restrict__ pW1,
        unsigned short* __restrict__ hseq) {
    extern __shared__ float smem[];
    float*          hp  = smem;                            // [64][768] fp32
    unsigned short* hbf = (unsigned short*)(hp + B_ * G3H_); // [64][256] bf16

    const int dir  = blockIdx.x;
    const float*          xp = dir ? xp1 : xp0;
    const unsigned short* pW = dir ? pW1 : pW0;

    const int tid  = threadIdx.x;
    const int lane = tid & 31;
    const int wv   = tid >> 5;          // 0..15
    const int mloc = lane & 15;
    const int kb   = (lane < 16) ? 0 : 8;

    float hreg[32];                     // fp32 h state, element i = tid + 512*n
#pragma unroll
    for (int n = 0; n < 32; ++n) hreg[n] = 0.f;
    for (int i = tid; i < B_ * H_; i += 512) hbf[i] = 0;
    __syncthreads();

    const v8f vzero = {0,0,0,0,0,0,0,0};

    for (int step = 0; step < T_; ++step) {
        const int t = dir ? (T_ - 1 - step) : step;

        // ---- WMMA phase: hp = h @ w_hh^T ----
        v8f acc[3][4];
#pragma unroll
        for (int nn = 0; nn < 3; ++nn)
#pragma unroll
            for (int mt = 0; mt < 4; ++mt) acc[nn][mt] = vzero;

#pragma unroll
        for (int kt = 0; kt < 8; ++kt) {
            FragU a[4];
#pragma unroll
            for (int mt = 0; mt < 4; ++mt) {
                const unsigned short* p =
                    hbf + (mt * 16 + mloc) * H_ + kt * 32 + kb;
                a[mt].u[0] = *(const uint4*)(p);
                a[mt].u[1] = *(const uint4*)(p + 16);
            }
#pragma unroll
            for (int nn = 0; nn < 3; ++nn) {
                const int nt = wv * 3 + nn;
                const unsigned short* pb = pW + ((long)nt * 8 + kt) * 512 + lane * 16;
                FragU b;
                b.u[0] = *(const uint4*)(pb);
                b.u[1] = *(const uint4*)(pb + 8);
#pragma unroll
                for (int mt = 0; mt < 4; ++mt)
                    acc[nn][mt] = __builtin_amdgcn_wmma_f32_16x16x32_bf16(
                        false, a[mt].v, false, b.v, (short)0, acc[nn][mt],
                        false, false);
            }
        }
        // scatter C fragments into hp (C layout: VGPR v -> M=v+8*(lane/16))
#pragma unroll
        for (int nn = 0; nn < 3; ++nn) {
            const int col = (wv * 3 + nn) * 16 + mloc;
#pragma unroll
            for (int mt = 0; mt < 4; ++mt) {
                const int rbase = mt * 16 + ((lane >> 4) << 3);
#pragma unroll
                for (int v = 0; v < 8; ++v)
                    hp[(rbase + v) * G3H_ + col] = acc[nn][mt][v];
            }
        }
        __syncthreads();

        // ---- gate math + state update ----
#pragma unroll
        for (int n = 0; n < 32; ++n) {
            const int  i   = tid + n * 512;
            const int  b   = i >> 8;
            const int  j   = i & 255;
            const long row = (long)b * T_ + t;
            const float xr = xp[row * G3H_ + j];
            const float xz = xp[row * G3H_ + H_ + j];
            const float xn = xp[row * G3H_ + 2 * H_ + j];
            const float hr = hp[b * G3H_ + j];
            const float hz = hp[b * G3H_ + H_ + j];
            const float hn = hp[b * G3H_ + 2 * H_ + j];
            const float r  = 1.f / (1.f + __expf(-(xr + hr)));
            const float z  = 1.f / (1.f + __expf(-(xz + hz)));
            const float ng = tanhf(xn + r * hn);
            const float hnew = (1.f - z) * ng + z * hreg[n];
            hreg[n] = hnew;
            const unsigned short hb = f2bfu(hnew);
            hbf[i] = hb;
            hseq[row * H2_ + dir * H_ + j] = hb;
        }
        __syncthreads();
    }
}

// ---------------------------------------------------------------------------
// z_tilde + KL reduction
// ---------------------------------------------------------------------------
__global__ void k_ztkl(const float* __restrict__ mu, const float* __restrict__ lv,
                       const float* __restrict__ eps, float* __restrict__ zt,
                       float* klacc) {
    long i = (long)blockIdx.x * blockDim.x + threadIdx.x;
    float kl = 0.f;
    if (i < (long)BT_ * NZ_) {
        float m = mu[i], l = lv[i];
        float el = __expf(l);
        zt[i] = m + __expf(0.5f * l) * eps[i];
        kl = -0.5f * (1.f + l - m * m - el);
    }
    for (int off = 16; off > 0; off >>= 1) kl += __shfl_down(kl, off, 32);
    if ((threadIdx.x & 31) == 0)
        atomicAdd(klacc, kl * (1.f / (float)BT_));
}

// ---------------------------------------------------------------------------
// Switch-network scan: one workgroup (128 threads) per batch element.
// ---------------------------------------------------------------------------
__global__ __launch_bounds__(128) void k_scan(
        const float* __restrict__ pre, const float* __restrict__ zt,
        const float* __restrict__ sw_w1, const float* __restrict__ sw_w2,
        const float* __restrict__ sw_b2, float* __restrict__ out) {
    __shared__ float w1zT[NZ_ * SWH_];   // [j][k], conflict-free in k
    __shared__ float w2s[SWH_];
    __shared__ float zsh[NZ_];
    __shared__ float red[SWH_];
    __shared__ float sBeta;

    const int b = blockIdx.x;
    const int k = threadIdx.x;
    for (int i = k; i < NZ_ * SWH_; i += 128) {
        int j = i / SWH_, kk = i % SWH_;
        w1zT[i] = sw_w1[kk * W1COLS_ + (NE_ + H2_) + j];
    }
    w2s[k] = sw_w2[k];
    if (k < NZ_) zsh[k] = 0.f;
    const float b2 = sw_b2[0];
    __syncthreads();

    const long BETA_OFF = (long)BT_ * NZ_ + 1;

    for (int t = 0; t < T_; ++t) {
        const long row = (long)b * T_ + t;
        float s = pre[row * SWH_ + k];
#pragma unroll 8
        for (int j = 0; j < NZ_; ++j) s += zsh[j] * w1zT[j * SWH_ + k];
        red[k] = fmaxf(s, 0.f) * w2s[k];
        __syncthreads();
        if (k < 32) {
            float p = red[k] + red[k + 32] + red[k + 64] + red[k + 96];
            for (int off = 16; off > 0; off >>= 1) p += __shfl_down(p, off, 32);
            if (k == 0) sBeta = 1.f / (1.f + __expf(-(p + b2)));
        }
        __syncthreads();
        const float beta = sBeta;
        if (k < NZ_) {
            float zn = beta * zt[row * NZ_ + k] + (1.f - beta) * zsh[k];
            zsh[k] = zn;
            out[row * NZ_ + k] = zn;               // z_seq at offset 0
        }
        if (k == 0) out[BETA_OFF + row] = beta;    // beta_seq
        __syncthreads();
    }
}

// ---------------------------------------------------------------------------
extern "C" void kernel_launch(void* const* d_in, const int* in_sizes, int n_in,
                              void* d_out, int out_size, void* d_ws, size_t ws_size,
                              hipStream_t stream) {
    const float* e_seq  = (const float*)d_in[0];
    const float* eps    = (const float*)d_in[1];
    const float* w_ih_f = (const float*)d_in[2];
    const float* w_hh_f = (const float*)d_in[3];
    const float* b_ih_f = (const float*)d_in[4];
    const float* b_hh_f = (const float*)d_in[5];
    const float* w_ih_b = (const float*)d_in[6];
    const float* w_hh_b = (const float*)d_in[7];
    const float* b_ih_b = (const float*)d_in[8];
    const float* b_hh_b = (const float*)d_in[9];
    const float* mu_w   = (const float*)d_in[10];
    const float* mu_b   = (const float*)d_in[11];
    const float* lv_w   = (const float*)d_in[12];
    const float* lv_b   = (const float*)d_in[13];
    const float* sw_w1  = (const float*)d_in[14];
    const float* sw_b1  = (const float*)d_in[15];
    const float* sw_w2  = (const float*)d_in[16];
    const float* sw_b2  = (const float*)d_in[17];

    char* ws = (char*)d_ws;
    size_t o = 0;
    auto alc = [&](size_t bytes) {
        size_t r = o;
        o += (bytes + 255) & ~(size_t)255;
        return r;
    };
    unsigned short* e_bf  = (unsigned short*)(ws + alc((size_t)BT_ * NE_ * 2));
    float*          xp0   = (float*)(ws + alc((size_t)BT_ * G3H_ * 4));
    float*          xp1   = (float*)(ws + alc((size_t)BT_ * G3H_ * 4));
    unsigned short* hseq  = (unsigned short*)(ws + alc((size_t)BT_ * H2_ * 2));
    float*          pre   = (float*)(ws + alc((size_t)BT_ * SWH_ * 4));
    float*          mu    = (float*)(ws + alc((size_t)BT_ * NZ_ * 4));
    float*          lv    = (float*)(ws + alc((size_t)BT_ * NZ_ * 4));
    float*          zt    = (float*)(ws + alc((size_t)BT_ * NZ_ * 4));
    unsigned short* pwihf = (unsigned short*)(ws + alc((size_t)G3H_ * NE_ * 2));
    unsigned short* pwihb = (unsigned short*)(ws + alc((size_t)G3H_ * NE_ * 2));
    unsigned short* pwhhf = (unsigned short*)(ws + alc((size_t)G3H_ * H_ * 2));
    unsigned short* pwhhb = (unsigned short*)(ws + alc((size_t)G3H_ * H_ * 2));
    unsigned short* pw1e  = (unsigned short*)(ws + alc((size_t)SWH_ * NE_ * 2));
    unsigned short* pw1h  = (unsigned short*)(ws + alc((size_t)SWH_ * H2_ * 2));
    unsigned short* pmuw  = (unsigned short*)(ws + alc((size_t)NZ_ * H2_ * 2));
    unsigned short* plvw  = (unsigned short*)(ws + alc((size_t)NZ_ * H2_ * 2));
    float*          biasf = (float*)(ws + alc((size_t)G3H_ * 4));
    float*          biasb = (float*)(ws + alc((size_t)G3H_ * 4));

    float* outp  = (float*)d_out;
    float* klptr = outp + (size_t)BT_ * NZ_;

    // zero KL accumulator (capture-safe)
    hipMemsetAsync((void*)klptr, 0, sizeof(float), stream);

    // e_seq -> bf16
    {
        long n = (long)BT_ * NE_;
        k_cvt_bf16<<<(unsigned)((n + 255) / 256), 256, 0, stream>>>(e_seq, e_bf, n);
    }
    // pack weights into B-fragment layout
    auto pack = [&](const float* W, int stride, int colOff, int N, int K,
                    unsigned short* dst) {
        long total = (long)(N / 16) * (K / 32) * 512;
        k_pack_b<<<(unsigned)((total + 255) / 256), 256, 0, stream>>>(
            W, stride, colOff, N, K, dst);
    };
    pack(w_ih_f, NE_, 0, G3H_, NE_, pwihf);
    pack(w_ih_b, NE_, 0, G3H_, NE_, pwihb);
    pack(w_hh_f, H_, 0, G3H_, H_, pwhhf);
    pack(w_hh_b, H_, 0, G3H_, H_, pwhhb);
    pack(sw_w1, W1COLS_, 0,         SWH_, NE_, pw1e);   // w1_e slice
    pack(sw_w1, W1COLS_, NE_,       SWH_, H2_, pw1h);   // w1_h slice
    pack(mu_w, H2_, 0, NZ_, H2_, pmuw);
    pack(lv_w, H2_, 0, NZ_, H2_, plvw);

    k_addvec<<<3, 256, 0, stream>>>(b_ih_f, b_hh_f, biasf, G3H_);
    k_addvec<<<3, 256, 0, stream>>>(b_ih_b, b_hh_b, biasb, G3H_);

    auto gemm = [&](const unsigned short* A, const unsigned short* pB,
                    const float* bias, const float* C0, float* out_, int N, int K) {
        int blocks = (BT_ / 16) * (N / 64) / 8;  // 8 waves per block
        k_gemm<<<blocks, 256, 0, stream>>>(A, pB, bias, C0, out_, N, K);
    };
    // input projections (b_hh folded into the bias)
    gemm(e_bf, pwihf, biasf, nullptr, xp0, G3H_, NE_);
    gemm(e_bf, pwihb, biasb, nullptr, xp1, G3H_, NE_);
    // pre_e = e @ w1_e^T + sw_b1
    gemm(e_bf, pw1e, sw_b1, nullptr, pre, SWH_, NE_);

    // GRU recurrence (both directions in parallel)
    hipFuncSetAttribute(reinterpret_cast<const void*>(k_gru),
                        hipFuncAttributeMaxDynamicSharedMemorySize, GRU_LDS);
    k_gru<<<2, 512, GRU_LDS, stream>>>(xp0, xp1, pwhhf, pwhhb, hseq);

    // head GEMMs on h_seq
    gemm(hseq, pmuw, mu_b, nullptr, mu, NZ_, H2_);
    gemm(hseq, plvw, lv_b, nullptr, lv, NZ_, H2_);
    gemm(hseq, pw1h, nullptr, pre, pre, SWH_, H2_);  // pre += h_seq @ w1_h^T

    // z_tilde + KL
    {
        long n = (long)BT_ * NZ_;
        k_ztkl<<<(unsigned)((n + 255) / 256), 256, 0, stream>>>(mu, lv, eps, zt, klptr);
    }
    // switch scan -> z_seq + beta_seq
    k_scan<<<B_, 128, 0, stream>>>(pre, zt, sw_w1, sw_w2, sw_b2, outp);
}
